// DepthEncoder_9_38989713113650
// MI455X (gfx1250) — matrix operand order repaired and checked by reading, not compiled, not run on hardware
//
#include <hip/hip_runtime.h>
#include <hip/hip_bf16.h>
#include <math.h>

// ---------------------------------------------------------------------------
// DepthEncoder for MI455X (gfx1250).  All heavy channel contractions are
// V_WMMA_F32_16X16X32_F16 GEMMs over N=2048 (=32 images * 64 positions).
// Intermediates are f16 [position][channel] matrices; epilogues fused.
// ---------------------------------------------------------------------------

typedef __attribute__((ext_vector_type(16))) _Float16 v16h;
typedef __attribute__((ext_vector_type(8)))  _Float16 v8h;
typedef __attribute__((ext_vector_type(8)))  float    v8f;

#define NTOT 2048          // B(32) * 8 * 8 spatial positions

// ---------------- input gather: nearest(input,8,8) -> f16 [2048][32] -------
__global__ void pack_low_k(const float* __restrict__ img, _Float16* __restrict__ act) {
  int idx = blockIdx.x * blockDim.x + threadIdx.x;      // 2048*32 threads
  int k = idx & 31, n = idx >> 5;
  int b = n >> 6, y = (n >> 3) & 7, x = n & 7;
  float v = 0.f;
  if (k < 16)  // iy = oy*256//8 = 32*y
    v = img[(((size_t)b * 16 + k) * 256 + (size_t)y * 32) * 256 + (size_t)x * 32];
  act[idx] = (_Float16)v;
}

// ---------------- f32 weight -> f16, with M/K zero padding -----------------
__global__ void cvt_w_k(const float* __restrict__ w, _Float16* __restrict__ o,
                        int Min, int Kin, int Mpad, int Kpad) {
  int idx = blockIdx.x * blockDim.x + threadIdx.x;
  if (idx >= Mpad * Kpad) return;
  int k = idx % Kpad, m = idx / Kpad;
  float v = (m < Min && k < Kin) ? w[(size_t)m * Kin + k] : 0.f;
  o[idx] = (_Float16)v;
}

// ---------------- BN folding: s = g*rsqrt(v+eps), t = b - m*s --------------
__global__ void bn_prep_k(const float* g, const float* bb, const float* mm,
                          const float* vv, float* s, float* t, int C) {
  int c = blockIdx.x * blockDim.x + threadIdx.x;
  if (c >= C) return;
  float sc = g[c] * rsqrtf(vv[c] + 1e-5f);
  s[c] = sc; t[c] = bb[c] - mm[c] * sc;
}

// ---------------- im2col 3x3 on 8x8, zero or reflect pad -------------------
__global__ void im2col_k(const _Float16* __restrict__ act, _Float16* __restrict__ col,
                         int C, int reflect) {
  int idx = blockIdx.x * blockDim.x + threadIdx.x;      // 2048*C threads
  int c = idx % C, n = idx / C;
  int b = n >> 6, y = (n >> 3) & 7, x = n & 7;
  _Float16* cp = col + (size_t)n * (9 * C) + (size_t)c * 9;
#pragma unroll
  for (int k = 0; k < 9; ++k) {
    int sy = y + k / 3 - 1, sx = x + k % 3 - 1;
    _Float16 v = (_Float16)0.f;
    if (reflect) {
      sy = sy < 0 ? -sy : (sy > 7 ? 14 - sy : sy);
      sx = sx < 0 ? -sx : (sx > 7 ? 14 - sx : sx);
      v = act[(size_t)((b << 6) + sy * 8 + sx) * C + c];
    } else if (sy >= 0 && sy < 8 && sx >= 0 && sx < 8) {
      v = act[(size_t)((b << 6) + sy * 8 + sx) * C + c];
    }
    cp[k] = v;
  }
}

// ---------------- depthwise kxk stride-k down conv -------------------------
__global__ void down_dw_k(const float* __restrict__ rgb, const float* __restrict__ w,
                          _Float16* __restrict__ rgbd, int C, int Hin, int kk) {
  int idx = blockIdx.x * blockDim.x + threadIdx.x;      // 32*C*64 threads
  int x = idx & 7, y = (idx >> 3) & 7;
  int c = (idx >> 6) % C, b = idx / (C << 6);
  const float* base = rgb + (((size_t)b * C + c) * Hin + (size_t)y * kk) * Hin + (size_t)x * kk;
  const float* wb   = w + (size_t)c * kk * kk;
  float s = 0.f;
  for (int dy = 0; dy < kk; ++dy)
    for (int dx = 0; dx < kk; ++dx)
      s += base[(size_t)dy * Hin + dx] * wb[dy * kk + dx];
  rgbd[(size_t)((b << 6) + y * 8 + x) * C + c] = (_Float16)s;
}

// ---------------- deformable bilinear gather -> im2col [n][c*9+k] ----------
__global__ void deform_sample_k(const _Float16* __restrict__ dact,
                                const float* __restrict__ off,
                                _Float16* __restrict__ col, int C) {
  int tid = blockIdx.x * blockDim.x + threadIdx.x;      // 2048*9 waves
  int gid = tid >> 5, lane = tid & 31;
  if (gid >= NTOT * 9) return;
  int n = gid / 9, k = gid - n * 9;
  int b = n >> 6, y = (n >> 3) & 7, x = n & 7;
  float dy = off[n * 18 + 2 * k], dx = off[n * 18 + 2 * k + 1];
  float py = (float)(y + k / 3 - 1) + dy;
  float px = (float)(x + k % 3 - 1) + dx;
  float y0f = floorf(py), x0f = floorf(px);
  float wy = py - y0f, wx = px - x0f;
  int y0 = (int)y0f, x0i = (int)x0f, y1 = y0 + 1, x1 = x0i + 1;
  float w00 = (1.f - wy) * (1.f - wx), w01 = (1.f - wy) * wx;
  float w10 = wy * (1.f - wx),         w11 = wy * wx;
  bool vy0 = (y0 >= 0) & (y0 < 8), vy1 = (y1 >= 0) & (y1 < 8);
  bool vx0 = (x0i >= 0) & (x0i < 8), vx1 = (x1 >= 0) & (x1 < 8);
  w00 = (vy0 && vx0) ? w00 : 0.f;  w01 = (vy0 && vx1) ? w01 : 0.f;
  w10 = (vy1 && vx0) ? w10 : 0.f;  w11 = (vy1 && vx1) ? w11 : 0.f;
  int yc0 = min(max(y0, 0), 7), yc1 = min(max(y1, 0), 7);
  int xc0 = min(max(x0i, 0), 7), xc1 = min(max(x1, 0), 7);
  const _Float16* p00 = dact + (size_t)((b << 6) + yc0 * 8 + xc0) * C;
  const _Float16* p01 = dact + (size_t)((b << 6) + yc0 * 8 + xc1) * C;
  const _Float16* p10 = dact + (size_t)((b << 6) + yc1 * 8 + xc0) * C;
  const _Float16* p11 = dact + (size_t)((b << 6) + yc1 * 8 + xc1) * C;
  _Float16* cp = col + (size_t)n * (9 * C) + k;
  for (int c = lane; c < C; c += 32) {
    float v = w00 * (float)p00[c] + w01 * (float)p01[c] +
              w10 * (float)p10[c] + w11 * (float)p11[c];
    cp[(size_t)c * 9] = (_Float16)v;
  }
}

// ---------------- nearest upsample f16[n][C] -> f32 NCHW -------------------
__global__ void upsample_k(const _Float16* __restrict__ feat, float* __restrict__ out,
                           int C, int Ho) {
  long long idx = (long long)blockIdx.x * blockDim.x + threadIdx.x;
  long long total = (long long)32 * C * Ho * Ho;
  if (idx >= total) return;
  int ox = (int)(idx % Ho); long long t = idx / Ho;
  int oy = (int)(t % Ho); t /= Ho;
  int c = (int)(t % C);  int b = (int)(t / C);
  int y = (oy * 8) / Ho, x = (ox * 8) / Ho;   // arange(Ho)*8//Ho
  out[idx] = (float)feat[(size_t)((b << 6) + y * 8 + x) * C + c];
}

// ---------------- WMMA GEMM: D[M x 2048] = W[M x K] * Act[2048 x K]^T ------
// One wave owns a 64(M) x 16(N) strip: 4 accumulators, 4 WMMAs per B frag.
// MODE 0: BN+ReLU -> f16 [n][M]        MODE 1: bias+ELU -> f16 [n][M]
// MODE 2: bias + residual -> f16       MODE 3: bias + clip(+-2) -> f32 [n][Mvalid]
template <int MODE>
__global__ void wmma_gemm_k(const _Float16* __restrict__ A,
                            const _Float16* __restrict__ Bm,
                            int M, int K,
                            const float* __restrict__ p0,
                            const float* __restrict__ p1,
                            const _Float16* __restrict__ resid,
                            _Float16* __restrict__ outH,
                            float* __restrict__ outF, int Mvalid) {
  int wid  = blockIdx.x * 8 + (threadIdx.x >> 5);
  int lane = threadIdx.x & 31;
  const int nTiles = NTOT >> 4;                 // 128
  int mBlocks = M >> 6;
  if (wid >= mBlocks * nTiles) return;
  int nb = wid % nTiles, mb = wid / nTiles;

  int l15  = lane & 15;
  int colN = nb * 16 + l15;
  int aK   = (lane < 16) ? 0 : 16;              // A: lanes 0-15 K{0-15}
  int bK   = (lane < 16) ? 0 : 16;              // B: lanes 0-15 K{0-15}

  const _Float16* bRow  = Bm + (size_t)colN * K + bK;
  const _Float16* aBase = A + (size_t)(mb * 64 + l15) * K + aK;

  v8f acc[4] = {};
  for (int k0 = 0; k0 < K; k0 += 32) {
    v16h bf = *(const v16h*)(bRow + k0);
#pragma unroll
    for (int t = 0; t < 4; ++t) {
      const _Float16* ap = aBase + (size_t)t * 16 * K + k0;
      v8h alo = *(const v8h*)(ap);
      v8h ahi = *(const v8h*)(ap + 8);
      v16h af = __builtin_shufflevector(alo, ahi,
                0, 1, 2, 3, 4, 5, 6, 7, 8, 9, 10, 11, 12, 13, 14, 15);
      acc[t] = __builtin_amdgcn_wmma_f32_16x16x32_f16(
                   false, af, false, bf, (short)0, acc[t], false, false);
    }
  }

  int mo = (lane >= 16) ? 8 : 0;
  int n  = colN;
#pragma unroll
  for (int t = 0; t < 4; ++t) {
    int mbase = mb * 64 + t * 16 + mo;
#pragma unroll
    for (int r = 0; r < 8; ++r) {
      int m = mbase + r;
      float v = acc[t][r];
      if (MODE == 0) {
        v = v * p0[m] + p1[m]; v = v > 0.f ? v : 0.f;
        outH[(size_t)n * M + m] = (_Float16)v;
      } else if (MODE == 1) {
        v += p0[m]; v = v > 0.f ? v : (__expf(v) - 1.f);
        outH[(size_t)n * M + m] = (_Float16)v;
      } else if (MODE == 2) {
        v += p0[m] + (float)resid[(size_t)n * M + m];
        outH[(size_t)n * M + m] = (_Float16)v;
      } else {
        v += p0[m]; v = fminf(fmaxf(v, -2.f), 2.f);
        if (m < Mvalid) outF[(size_t)n * Mvalid + m] = v;
      }
    }
  }
}

// ---------------------------------------------------------------------------
extern "C" void kernel_launch(void* const* d_in, const int* in_sizes, int n_in,
                              void* d_out, int out_size, void* d_ws, size_t ws_size,
                              hipStream_t stream) {
  (void)in_sizes; (void)n_in; (void)out_size; (void)ws_size;
  static const int CH[5] = {64, 64, 128, 256, 512};

  // ---- workspace carve (~31 MB, all 256B aligned) ----
  char* w = (char*)d_ws; size_t o = 0;
  auto carve = [&](size_t bytes) { void* p = w + o; o = (o + bytes + 255) & ~(size_t)255; return p; };
  _Float16* actA = (_Float16*)carve((size_t)NTOT * 512 * 2);
  _Float16* actB = (_Float16*)carve((size_t)NTOT * 512 * 2);
  _Float16* dbuf = (_Float16*)carve((size_t)NTOT * 512 * 2);
  _Float16* colb = (_Float16*)carve((size_t)NTOT * 4608 * 2);
  _Float16* rgbd = (_Float16*)carve((size_t)NTOT * 512 * 2);
  float*    offb = (float*)carve((size_t)NTOT * 18 * 4);
  _Float16* wbuf = (_Float16*)carve((size_t)512 * 4608 * 2);
  float*    bns  = (float*)carve(512 * 4);
  float*    bnt  = (float*)carve(512 * 4);

  auto cvt = [&](const float* src, int Min, int Kin, int Mpad, int Kpad) {
    int tot = Mpad * Kpad;
    cvt_w_k<<<(tot + 255) / 256, 256, 0, stream>>>(src, wbuf, Min, Kin, Mpad, Kpad);
  };
  auto gemm = [&](int mode, int M, int K, const _Float16* B,
                  const float* p0, const float* p1, const _Float16* resid,
                  _Float16* oH, float* oF, int Mvalid) {
    int blocks = (M >> 6) * 16;   // ((M/64)*128 waves) / 8 waves per block
    switch (mode) {
      case 0: wmma_gemm_k<0><<<blocks, 256, 0, stream>>>(wbuf, B, M, K, p0, p1, resid, oH, oF, Mvalid); break;
      case 1: wmma_gemm_k<1><<<blocks, 256, 0, stream>>>(wbuf, B, M, K, p0, p1, resid, oH, oF, Mvalid); break;
      case 2: wmma_gemm_k<2><<<blocks, 256, 0, stream>>>(wbuf, B, M, K, p0, p1, resid, oH, oF, Mvalid); break;
      default: wmma_gemm_k<3><<<blocks, 256, 0, stream>>>(wbuf, B, M, K, p0, p1, resid, oH, oF, Mvalid); break;
    }
  };

  // ---- input gather ----
  pack_low_k<<<(NTOT * 32) / 256, 256, 0, stream>>>((const float*)d_in[0], actA);

  const _Float16* actPrev = actA;
  size_t outOfs = 0;

  for (int i = 0; i < 5; ++i) {
    int C = CH[i];
    const float* rgb   = (const float*)d_in[1 + i];
    const float* dc_w  = (const float*)d_in[6 + i];
    const float* bn_g  = (const float*)d_in[11 + 4 * i + 0];
    const float* bn_b  = (const float*)d_in[11 + 4 * i + 1];
    const float* bn_m  = (const float*)d_in[11 + 4 * i + 2];
    const float* bn_v  = (const float*)d_in[11 + 4 * i + 3];
    const float* dwn_w = (const float*)d_in[39 + i];
    const float* off_w = (const float*)d_in[44 + i];
    const float* off_b = (const float*)d_in[49 + i];
    const float* def_w = (const float*)d_in[54 + i];
    const float* def_b = (const float*)d_in[59 + i];

    // --- 1x1 conv + BN + ReLU -> dbuf ---
    int Kreal = (i == 0) ? 16 : CH[i - 1];
    int Kpad  = (i == 0) ? 32 : Kreal;           // K must be a multiple of 32
    cvt(dc_w, C, Kreal, C, Kpad);
    bn_prep_k<<<(C + 255) / 256, 256, 0, stream>>>(bn_g, bn_b, bn_m, bn_v, bns, bnt, C);
    gemm(0, C, Kpad, actPrev, bns, bnt, nullptr, dbuf, nullptr, C);

    // --- reflect-pad 3x3 conv + ELU (levels 1..4), in place on dbuf ---
    if (i > 0) {
      const float* cb_w = (const float*)d_in[31 + (i - 1)];
      const float* cb_b = (const float*)d_in[35 + (i - 1)];
      im2col_k<<<(NTOT * C) / 256, 256, 0, stream>>>(dbuf, colb, C, 1);
      cvt(cb_w, C, 9 * C, C, 9 * C);
      gemm(1, C, 9 * C, colb, cb_b, nullptr, nullptr, dbuf, nullptr, C);
    }

    // --- depthwise stride-k down conv (reads full rgb pyramid level) ---
    int Hin = 128 >> i, kk = Hin / 8;
    down_dw_k<<<(32 * C * 64) / 256, 256, 0, stream>>>(rgb, dwn_w, rgbd, C, Hin, kk);

    // --- offset conv: zero-pad im2col + GEMM (M padded 18 -> 64), clip +-2 ---
    im2col_k<<<(NTOT * C) / 256, 256, 0, stream>>>(rgbd, colb, C, 0);
    cvt(off_w, 18, 9 * C, 64, 9 * C);
    gemm(3, 64, 9 * C, colb, off_b, nullptr, nullptr, nullptr, offb, 18);

    // --- deformable conv: bilinear gather -> im2col, GEMM + bias + residual ---
    deform_sample_k<<<(NTOT * 9 * 32) / 256, 256, 0, stream>>>(dbuf, offb, colb, C);
    cvt(def_w, C, 9 * C, C, 9 * C);
    _Float16* feat = (i % 2 == 0) ? actB : actA;
    gemm(2, C, 9 * C, colb, def_b, nullptr, dbuf, feat, nullptr, C);

    // --- nearest upsample to output slice ---
    int Ho = 128 >> i;
    long long tot = (long long)32 * C * Ho * Ho;
    upsample_k<<<(unsigned)((tot + 255) / 256), 256, 0, stream>>>(feat, (float*)d_out + outOfs, C, Ho);
    outOfs += (size_t)tot;

    actPrev = feat;
  }
}
